// MultiHeadAttention_30872224924261
// MI455X (gfx1250) — matrix operand order repaired and checked by reading, not compiled, not run on hardware
//
#include <hip/hip_runtime.h>
#include <hip/hip_bf16.h>

// Problem constants (from reference): B=4, S=2048, D=1024, H=16, DK=64
#define BB 4
#define SS 2048
#define DD 1024
#define HH 16
#define DKK 64
#define MM (BB * SS)   // 8192 rows in all projection GEMMs

#define LOG2E 1.4426950408889634f

typedef __attribute__((ext_vector_type(16))) __bf16 v16bf;
typedef __attribute__((ext_vector_type(8)))  float v8f;

__device__ __forceinline__ unsigned short f2bf(float x) {
  union { __bf16 b; unsigned short u; } t;
  t.b = (__bf16)x;
  return t.u;
}
// bf16 (as packed u32) -> f32 decodes: low half and high half of a dword.
__device__ __forceinline__ float bflo(unsigned u) {
  union { unsigned x; float f; } t; t.x = u << 16; return t.f;
}
__device__ __forceinline__ float bfhi(unsigned u) {
  union { unsigned x; float f; } t; t.x = u & 0xffff0000u; return t.f;
}

// Raw v_exp_f32 (exp2). All softmax math is kept in log2 space.
__device__ __forceinline__ float fast_exp2(float x) {
#if __has_builtin(__builtin_amdgcn_exp2f)
  return __builtin_amdgcn_exp2f(x);
#else
  return exp2f(x);
#endif
}

__device__ __forceinline__ v8f wmma_bf16(v16bf a, v16bf b, v8f c) {
  // D = A(16x32 bf16) x B(32x16 bf16) + C(16x16 f32)
  return __builtin_amdgcn_wmma_f32_16x16x32_bf16(
      false, a, false, b, (short)0, c, false, false);
}

// Load 16 bf16 (two 16B chunks) from memory (global or LDS) into an A/B fragment.
__device__ __forceinline__ v16bf load_16h(const unsigned short* p0,
                                          const unsigned short* p1) {
  union { uint4 u[2]; v16bf v; } t;
  t.u[0] = *(const uint4*)p0;
  t.u[1] = *(const uint4*)p1;
  return t.v;
}

// A-fragment (16x32, bf16) from row-major bf16 src with row stride ld.
// Lanes 0-15: row=lane, K={kb..kb+7, kb+16..kb+23};
// lanes 16-31: row=lane-16, K={kb+8..kb+15, kb+24..kb+31}.
__device__ __forceinline__ v16bf load_a_h(const unsigned short* src, int m0,
                                          int kb, int ld, int lane) {
  int row = m0 + (lane & 15);
  int lo = (lane < 16) ? 0 : 8;
  const unsigned short* p = src + (size_t)row * ld + kb + lo;
  return load_16h(p, p + 16);
}

// A-fragment from row-major fp32 src (converted to bf16 in-register).
__device__ __forceinline__ v16bf load_a_f32(const float* src, int m0, int kb,
                                            int ld, int lane) {
  int row = m0 + (lane & 15);
  int lo = (lane < 16) ? 0 : 8;
  const float* p0 = src + (size_t)row * ld + kb + lo;
  const float* p1 = p0 + 16;
  float4 f0 = *(const float4*)(p0);
  float4 f1 = *(const float4*)(p0 + 4);
  float4 f2 = *(const float4*)(p1);
  float4 f3 = *(const float4*)(p1 + 4);
  v16bf v;
  v[0]  = (__bf16)f0.x; v[1]  = (__bf16)f0.y; v[2]  = (__bf16)f0.z; v[3]  = (__bf16)f0.w;
  v[4]  = (__bf16)f1.x; v[5]  = (__bf16)f1.y; v[6]  = (__bf16)f1.z; v[7]  = (__bf16)f1.w;
  v[8]  = (__bf16)f2.x; v[9]  = (__bf16)f2.y; v[10] = (__bf16)f2.z; v[11] = (__bf16)f2.w;
  v[12] = (__bf16)f3.x; v[13] = (__bf16)f3.y; v[14] = (__bf16)f3.z; v[15] = (__bf16)f3.w;
  return v;
}

// B-fragment (32x16, bf16). src stores element (k, n) at src[n*ld + k]
// (N-major). Lanes 0-15: col n0+lane, K={kb..kb+7, kb+16..kb+23}; lanes 16-31: K={kb+8..kb+15, kb+24..kb+31}.
__device__ __forceinline__ v16bf load_b(const unsigned short* src, int n0,
                                        int kb, int ld, int lane) {
  int col = n0 + (lane & 15);
  int k = kb + ((lane < 16) ? 0 : 8);
  const unsigned short* p = src + (size_t)col * ld + k;
  return load_16h(p, p + 16);
}

// ---------------------------------------------------------------------------
// 16-lane butterfly reductions done entirely with DPP (no LDS pipe):
// quad_perm(1,0,3,2)=xor1, quad_perm(2,3,0,1)=xor2, row_half_mirror=lane^7,
// row_mirror=lane^15. A C-fragment row lives in one VGPR slot across a
// 16-lane half, so this reduces exactly one row; both halves independent.
// ---------------------------------------------------------------------------
template <int CTRL>
__device__ __forceinline__ float dpp_mov(float x) {
  union { float f; int i; } a, r;
  a.f = x;
  r.i = __builtin_amdgcn_update_dpp(0, a.i, CTRL, 0xf, 0xf, true);
  return r.f;
}
__device__ __forceinline__ float hmax16(float x) {
  x = fmaxf(x, dpp_mov<0xB1>(x));   // quad_perm(1,0,3,2)
  x = fmaxf(x, dpp_mov<0x4E>(x));   // quad_perm(2,3,0,1)
  x = fmaxf(x, dpp_mov<0x141>(x));  // row_half_mirror
  x = fmaxf(x, dpp_mov<0x140>(x));  // row_mirror
  return x;
}
__device__ __forceinline__ float hsum16(float x) {
  x += dpp_mov<0xB1>(x);
  x += dpp_mov<0x4E>(x);
  x += dpp_mov<0x141>(x);
  x += dpp_mov<0x140>(x);
  return x;
}

// ---------------------------------------------------------------------------
// Weight transpose + fp32->bf16 convert: W[k][n] (row-major KxN) -> Wt[n][k]
// ---------------------------------------------------------------------------
__global__ void __launch_bounds__(256) wt_cvt_kernel(
    const float* __restrict__ W, unsigned short* __restrict__ Wt) {
  __shared__ float tile[32][33];
  int tx = threadIdx.x & 31;
  int ty = threadIdx.x >> 5;  // 0..7
  int n0 = blockIdx.x * 32;
  int k0 = blockIdx.y * 32;
  #pragma unroll
  for (int i = ty; i < 32; i += 8)
    tile[i][tx] = W[(size_t)(k0 + i) * DD + n0 + tx];
  __syncthreads();
  #pragma unroll
  for (int i = ty; i < 32; i += 8)
    Wt[(size_t)(n0 + i) * DD + k0 + tx] = f2bf(tile[tx][i]);
}

// ---------------------------------------------------------------------------
// Mask transpose + convert to additive bf16 bias (log2 space):
// biasT[b][key][q] = mask[b][q][key] ? 0 : -1e9*log2(e)
// Attention reads 8 consecutive q per lane -> one 16B load per fragment.
// ---------------------------------------------------------------------------
__global__ void __launch_bounds__(256) mask_cvt_kernel(
    const int* __restrict__ mask, unsigned short* __restrict__ biasT) {
  __shared__ int tile[32][33];
  int tx = threadIdx.x & 31;
  int ty = threadIdx.x >> 5;
  int k0 = blockIdx.x * 32;  // key
  int q0 = blockIdx.y * 32;  // query
  int b = blockIdx.z;
  const int* mb = mask + (size_t)b * SS * SS;
  unsigned short* ob = biasT + (size_t)b * SS * SS;
  const unsigned short NEG = f2bf(-1e9f * LOG2E);
  #pragma unroll
  for (int i = ty; i < 32; i += 8)
    tile[i][tx] = mb[(size_t)(q0 + i) * SS + k0 + tx];
  __syncthreads();
  #pragma unroll
  for (int i = ty; i < 32; i += 8)
    ob[(size_t)(k0 + i) * SS + q0 + tx] =
        tile[tx][i] ? (unsigned short)0 : NEG;
}

// ---------------------------------------------------------------------------
// Projection GEMM: Y = (X @ W + b) * oscale   (M=8192, N=1024, K=1024)
// MODE 0: fp32 X, bf16 out head-split [b,h,s,dk]        (Q, K projections)
// MODE 1: fp32 X, bf16 out transposed [b,h,dk,s]        (V projection)
// MODE 2: bf16 X (concat), fp32 out row-major [m,n]     (output projection)
// Block: 256 threads = 8 waves (4 M x 2 N); wave tile 32x64; K step 32.
// ---------------------------------------------------------------------------
template <int MODE>
__global__ void __launch_bounds__(256, 1) proj_gemm_kernel(
    const float* __restrict__ Xf, const unsigned short* __restrict__ Xh,
    const unsigned short* __restrict__ Wt, const float* __restrict__ bias,
    float oscale, unsigned short* __restrict__ outh,
    float* __restrict__ outf) {
  const int lane = threadIdx.x & 31;
  const int w = threadIdx.x >> 5;
  const int wm = w & 3;
  const int wn = w >> 2;
  const int m0 = blockIdx.y * 128 + wm * 32;
  const int n0 = blockIdx.x * 128 + wn * 64;

  v8f acc[2][4] = {};

  for (int kb = 0; kb < DD; kb += 32) {
    v16bf a0, a1;
    if (MODE == 2) {
      a0 = load_a_h(Xh, m0, kb, DD, lane);
      a1 = load_a_h(Xh, m0 + 16, kb, DD, lane);
    } else {
      a0 = load_a_f32(Xf, m0, kb, DD, lane);
      a1 = load_a_f32(Xf, m0 + 16, kb, DD, lane);
    }
    v16bf bf[4];
    #pragma unroll
    for (int j = 0; j < 4; ++j) bf[j] = load_b(Wt, n0 + 16 * j, kb, DD, lane);
    #pragma unroll
    for (int j = 0; j < 4; ++j) {
      acc[0][j] = wmma_bf16(a0, bf[j], acc[0][j]);
      acc[1][j] = wmma_bf16(a1, bf[j], acc[1][j]);
    }
  }

  // Epilogue: bias add, optional scale, layout-specific store.
  const int mh = (lane < 16) ? 0 : 8;  // C layout: lanes 16-31 hold rows r+8
  #pragma unroll
  for (int j = 0; j < 4; ++j) {
    const int n = n0 + 16 * j + (lane & 15);
    const float bj = bias[n];
    #pragma unroll
    for (int i = 0; i < 2; ++i) {
      const int mbase = m0 + 16 * i + mh;  // 8-aligned; rows mbase..mbase+7
      if (MODE == 1) {
        // [b,h,dk,s]: fixed dk per lane, s contiguous over r -> one 16B store
        const int bi = mbase >> 11;
        const int s = mbase & (SS - 1);
        const int h = n >> 6;
        const int dk = n & (DKK - 1);
        union { uint4 u; unsigned short h8[8]; } pk;
        #pragma unroll
        for (int r = 0; r < 8; ++r)
          pk.h8[r] = f2bf((acc[i][j][r] + bj) * oscale);
        *(uint4*)(outh + (((size_t)bi * HH + h) * DKK + dk) * SS + s) = pk.u;
      } else {
        #pragma unroll
        for (int r = 0; r < 8; ++r) {
          const int m = mbase + r;
          const float val = (acc[i][j][r] + bj) * oscale;
          if (MODE == 2) {
            outf[(size_t)m * DD + n] = val;
          } else {
            const int bi = m >> 11;
            const int s = m & (SS - 1);
            const int h = n >> 6;
            const int dk = n & (DKK - 1);
            outh[(((size_t)bi * HH + h) * SS + s) * DKK + dk] = f2bf(val);
          }
        }
      }
    }
  }
}

// ---------------------------------------------------------------------------
// Flash attention: per wave, 16 queries, stream over 64 key tiles of 32 with
// online softmax in log2 space; fp32 accumulation; bf16 WMMA for QK^T, P·V.
// Qh: [b,h,s,dk] bf16 (pre-scaled by log2(e)/sqrt(DK)); Kh: [b,h,s,dk] bf16;
// Vt: [b,h,dk,s] bf16; biasT: [b,key,q] bf16 additive mask bias (log2 space).
// Output: concat [b,s,D] bf16.
// ---------------------------------------------------------------------------
__global__ void __launch_bounds__(256, 1) attn_kernel(
    const unsigned short* __restrict__ Qh, const unsigned short* __restrict__ Kh,
    const unsigned short* __restrict__ Vt,
    const unsigned short* __restrict__ biasT,
    unsigned short* __restrict__ Oc) {
  // Per-wave P staging tile: 16 rows x 32 cols bf16, row stride 40 halfs
  // (80B) so A-fragment chunk reads stay 16B aligned and banks decorrelate.
  __shared__ __align__(16) unsigned short ptile[8][16 * 40];

  const int lane = threadIdx.x & 31;
  const int w = threadIdx.x >> 5;
  const int bh = blockIdx.x >> 4;          // 16 query-blocks per (b,h)
  const int qblk = blockIdx.x & 15;
  const int b = bh >> 4;                   // bh / H
  const int h = bh & (HH - 1);
  const int q0 = qblk * 128 + w * 16;

  const unsigned short* Q = Qh + (size_t)bh * SS * DKK;   // [s][dk]
  const unsigned short* K = Kh + (size_t)bh * SS * DKK;   // [s][dk]
  const unsigned short* V = Vt + (size_t)bh * DKK * SS;   // [dk][s]
  const unsigned short* Bm = biasT + (size_t)b * SS * SS; // [key][q]

  const int lo = (lane < 16) ? 0 : 8;
  const int kc = (lane & 15);
  const int qrow_base = q0 + ((lane < 16) ? 0 : 8);

  // Q A-fragments held in registers for the whole key loop: dk 0..31, 32..63
  const unsigned short* qrp = Q + (size_t)(q0 + kc) * DKK;
  const v16bf qa0 = load_16h(qrp + lo, qrp + 16 + lo);
  const v16bf qa1 = load_16h(qrp + 32 + lo, qrp + 48 + lo);

  v8f o[4] = {};
  float mrow[8], lrow[8];
  #pragma unroll
  for (int r = 0; r < 8; ++r) { mrow[r] = -1e30f; lrow[r] = 0.0f; }

  unsigned short* pt = ptile[w];
  // bias pointer walks down the [key][q] table with the key tile
  const unsigned short* bp = Bm + (size_t)kc * SS + qrow_base;

  for (int k0 = 0; k0 < SS; k0 += 32) {
    __builtin_prefetch(K + (size_t)(k0 + 32) * DKK, 0, 1);
    __builtin_prefetch(V + (size_t)(k0 + 32), 0, 1);

    // --- scores: S = Q(16x64) x K^T(64x32), two 16-key groups ---
    v16bf kb00 = load_b(K, k0, 0, DKK, lane);       // keys k0..+15, dk 0..31
    v16bf kb01 = load_b(K, k0, 32, DKK, lane);      // keys k0..+15, dk 32..63
    v16bf kb10 = load_b(K, k0 + 16, 0, DKK, lane);  // keys +16..+31
    v16bf kb11 = load_b(K, k0 + 16, 32, DKK, lane);
    v8f s0 = {}, s1 = {};
    s0 = wmma_bf16(qa0, kb00, s0);
    s0 = wmma_bf16(qa1, kb01, s0);
    s1 = wmma_bf16(qa0, kb10, s1);
    s1 = wmma_bf16(qa1, kb11, s1);

    // --- additive mask bias: one 16B load per score fragment ---
    uint4 ub0 = *(const uint4*)(bp + (size_t)k0 * SS);
    uint4 ub1 = *(const uint4*)(bp + (size_t)(k0 + 16) * SS);
    s0[0] += bflo(ub0.x); s0[1] += bfhi(ub0.x);
    s0[2] += bflo(ub0.y); s0[3] += bfhi(ub0.y);
    s0[4] += bflo(ub0.z); s0[5] += bfhi(ub0.z);
    s0[6] += bflo(ub0.w); s0[7] += bfhi(ub0.w);
    s1[0] += bflo(ub1.x); s1[1] += bfhi(ub1.x);
    s1[2] += bflo(ub1.y); s1[3] += bfhi(ub1.y);
    s1[4] += bflo(ub1.z); s1[5] += bfhi(ub1.z);
    s1[6] += bflo(ub1.w); s1[7] += bfhi(ub1.w);

    // --- online softmax update (log2 space, raw v_exp_f32) ---
    #pragma unroll
    for (int r = 0; r < 8; ++r) {
      float t = hmax16(fmaxf(s0[r], s1[r]));
      const float mn = fmaxf(mrow[r], t);
      const float alpha = fast_exp2(mrow[r] - mn);
      mrow[r] = mn;
      s0[r] = fast_exp2(s0[r] - mn);
      s1[r] = fast_exp2(s1[r] - mn);
      lrow[r] = lrow[r] * alpha + hsum16(s0[r] + s1[r]);
      o[0][r] *= alpha; o[1][r] *= alpha; o[2][r] *= alpha; o[3][r] *= alpha;
    }

    // --- reshape P (C layout) -> A-fragment layout via wave-private LDS ---
    const int pr_base = (lane < 16) ? 0 : 8;
    #pragma unroll
    for (int r = 0; r < 8; ++r) {
      const int pr = pr_base + r;
      pt[pr * 40 + kc] = f2bf(s0[r]);
      pt[pr * 40 + 16 + kc] = f2bf(s1[r]);
    }
    const unsigned short* prp = pt + kc * 40;
    const v16bf pa = load_16h(prp + lo, prp + 16 + lo);

    // --- O += P(16x32) x V(32x64) ---
    #pragma unroll
    for (int j = 0; j < 4; ++j) {
      v16bf vb = load_b(V, j * 16, k0, SS, lane);  // cols dk=j*16.., keys k0..+31
      o[j] = wmma_bf16(pa, vb, o[j]);
    }
  }

  // --- normalize and store to concat [b,s,D] bf16 ---
  #pragma unroll
  for (int r = 0; r < 8; ++r) {
    const float inv = 1.0f / lrow[r];
    const int qm = qrow_base + r;
    const size_t base = ((size_t)b * SS + qm) * DD + h * DKK + kc;
    Oc[base]      = f2bf(o[0][r] * inv);
    Oc[base + 16] = f2bf(o[1][r] * inv);
    Oc[base + 32] = f2bf(o[2][r] * inv);
    Oc[base + 48] = f2bf(o[3][r] * inv);
  }
}

// ---------------------------------------------------------------------------
extern "C" void kernel_launch(void* const* d_in, const int* in_sizes, int n_in,
                              void* d_out, int out_size, void* d_ws,
                              size_t ws_size, hipStream_t stream) {
  const float* q = (const float*)d_in[0];
  const float* k = (const float*)d_in[1];
  const float* v = (const float*)d_in[2];
  const int* mask = (const int*)d_in[3];
  const float* Wq = (const float*)d_in[4];
  const float* bq = (const float*)d_in[5];
  const float* Wk = (const float*)d_in[6];
  const float* bk = (const float*)d_in[7];
  const float* Wv = (const float*)d_in[8];
  const float* bv = (const float*)d_in[9];
  const float* Wo = (const float*)d_in[10];
  const float* bo = (const float*)d_in[11];

  char* ws = (char*)d_ws;
  const size_t MB = 1024 * 1024;
  unsigned short* Wtq = (unsigned short*)(ws + 0 * MB);   // 2 MB each
  unsigned short* Wtk = (unsigned short*)(ws + 2 * MB);
  unsigned short* Wtv = (unsigned short*)(ws + 4 * MB);
  unsigned short* Wto = (unsigned short*)(ws + 6 * MB);
  unsigned short* QhB = (unsigned short*)(ws + 8 * MB);   // 16 MB each
  unsigned short* KhB = (unsigned short*)(ws + 24 * MB);
  unsigned short* VtB = (unsigned short*)(ws + 40 * MB);
  unsigned short* CcB = (unsigned short*)(ws + 56 * MB);  // concat, 16 MB
  unsigned short* MbB = (unsigned short*)(ws + 72 * MB);  // mask bias, 32 MB

  dim3 tb(256);

  // 1) weight transpose + bf16 convert; mask -> transposed bf16 bias
  dim3 tg(DD / 32, DD / 32);
  wt_cvt_kernel<<<tg, tb, 0, stream>>>(Wq, Wtq);
  wt_cvt_kernel<<<tg, tb, 0, stream>>>(Wk, Wtk);
  wt_cvt_kernel<<<tg, tb, 0, stream>>>(Wv, Wtv);
  wt_cvt_kernel<<<tg, tb, 0, stream>>>(Wo, Wto);
  dim3 mg(SS / 32, SS / 32, BB);
  mask_cvt_kernel<<<mg, tb, 0, stream>>>(mask, MbB);

  // 2) Q/K/V projections (WMMA bf16, fp32 accumulate);
  //    Q pre-scaled by log2(e)/sqrt(DK) so softmax runs on raw v_exp_f32.
  dim3 pg(DD / 128, MM / 128);  // (8, 64)
  proj_gemm_kernel<0><<<pg, tb, 0, stream>>>(q, nullptr, Wtq, bq,
                                             0.125f * LOG2E, QhB, nullptr);
  proj_gemm_kernel<0><<<pg, tb, 0, stream>>>(k, nullptr, Wtk, bk, 1.0f, KhB, nullptr);
  proj_gemm_kernel<1><<<pg, tb, 0, stream>>>(v, nullptr, Wtv, bv, 1.0f, VtB, nullptr);

  // 3) flash attention (1024 blocks x 8 waves x 16 queries)
  attn_kernel<<<BB * HH * (SS / 128), tb, 0, stream>>>(QhB, KhB, VtB, MbB, CcB);

  // 4) output projection -> fp32 d_out
  proj_gemm_kernel<2><<<pg, tb, 0, stream>>>(nullptr, CcB, Wto, bo, 1.0f,
                                             nullptr, (float*)d_out);
}
